// MolecularE3nnTransformerUpdate_13932873909295
// MI455X (gfx1250) — matrix operand order repaired and checked by reading, not compiled, not run on hardware
//
#include <hip/hip_runtime.h>
#include <math.h>

// Problem constants (match reference)
#define Nn   50000
#define Ee   800000
#define CCH  8
#define Gg   64
#define HIDc 128
#define EPSc 1e-5f

typedef __attribute__((ext_vector_type(16))) _Float16 v16h;
typedef __attribute__((ext_vector_type(8)))  _Float16 v8h;
typedef __attribute__((ext_vector_type(8)))  float    v8f;
typedef __attribute__((ext_vector_type(4)))  unsigned int v4u;
typedef __attribute__((ext_vector_type(8)))  int      v8i;
typedef __attribute__((ext_vector_type(4)))  int      v4i;

// Tensor Data Mover availability (probe-verified builtin names; arity differs by toolchain)
#if defined(__has_builtin)
#if __has_builtin(__builtin_amdgcn_tensor_load_to_lds) && __has_builtin(__builtin_amdgcn_s_wait_tensorcnt)
#define USE_TDM 1
#endif
#endif
#ifndef USE_TDM
#define USE_TDM 0
#endif

// ---------------------------------------------------------------- utilities
__global__ void k_zero(float* __restrict__ p, long n) {
  long i  = (long)blockIdx.x * blockDim.x + threadIdx.x;
  long st = (long)gridDim.x * blockDim.x;
  for (; i < n; i += st) p[i] = 0.f;
}

// ------------------------------------------- edge spherical harmonics + scatter
__global__ void k_edge_sh(const float* __restrict__ vec, const int* __restrict__ dst,
                          float* __restrict__ node_sh, float* __restrict__ cnt) {
  int e = blockIdx.x * blockDim.x + threadIdx.x;
  if (e >= Ee) return;
  float x = vec[e * 3 + 0], y = vec[e * 3 + 1], z = vec[e * 3 + 2];
  float r = sqrtf(x * x + y * y + z * z + 1e-12f);
  float inv = 1.f / r;
  x *= inv; y *= inv; z *= inv;
  const float s3 = 1.7320508075688772f, s15 = 3.872983346207417f, s5 = 2.23606797749979f;
  float sh[9];
  sh[0] = 1.f;
  sh[1] = s3 * x; sh[2] = s3 * y; sh[3] = s3 * z;
  sh[4] = s15 * x * z;
  sh[5] = s15 * x * y;
  sh[6] = s5 * (y * y - 0.5f * (x * x + z * z));
  sh[7] = s15 * y * z;
  sh[8] = 0.5f * s15 * (z * z - x * x);
  int d = dst[e];
#pragma unroll
  for (int j = 0; j < 9; ++j) atomicAdd(&node_sh[(long)d * 9 + j], sh[j]);
  atomicAdd(&cnt[d], 1.f);
}

// ------------------------------------------------------- node feature init
__global__ void k_node_init(const float* __restrict__ xs, const float* __restrict__ node_sh,
                            const float* __restrict__ cnt, float* __restrict__ sA,
                            float* __restrict__ vA, float* __restrict__ dA) {
  int n = blockIdx.x * blockDim.x + threadIdx.x;
  if (n >= Nn) return;
  float inv = 1.f / fmaxf(cnt[n], 1.f);
#pragma unroll
  for (int j = 0; j < 9; ++j) sA[(long)n * 16 + j] = xs[(long)n * 9 + j];
  sA[(long)n * 16 + 9] = node_sh[(long)n * 9 + 0] * inv;
#pragma unroll
  for (int j = 10; j < 16; ++j) sA[(long)n * 16 + j] = 0.f;
#pragma unroll
  for (int m = 0; m < 3; ++m) vA[(long)n * 24 + m] = node_sh[(long)n * 9 + 1 + m] * inv;
#pragma unroll
  for (int m = 0; m < 5; ++m) dA[(long)n * 40 + m] = node_sh[(long)n * 9 + 4 + m] * inv;
}

// -------------------------------------- o3_linear + norm_act for q,k,v (per node)
__global__ void k_qkv(const float* __restrict__ sA, const float* __restrict__ vA,
                      const float* __restrict__ dA, const float* __restrict__ W0,
                      const float* __restrict__ W1, const float* __restrict__ W2,
                      int cin_s, int cin_v,
                      float* __restrict__ qs, float* __restrict__ qv, float* __restrict__ qd,
                      float* __restrict__ ks, float* __restrict__ kv, float* __restrict__ kd,
                      float* __restrict__ vs, float* __restrict__ vv, float* __restrict__ vd) {
  int n = blockIdx.x * blockDim.x + threadIdx.x;
  if (n >= Nn) return;
  float si[10];
  for (int i = 0; i < cin_s; ++i) si[i] = sA[(long)n * 16 + i];
  float vi[CCH][3], di[CCH][5];
  for (int i = 0; i < cin_v; ++i) {
    for (int m = 0; m < 3; ++m) vi[i][m] = vA[(long)n * 24 + i * 3 + m];
    for (int m = 0; m < 5; ++m) di[i][m] = dA[(long)n * 40 + i * 5 + m];
  }
  float* outS[3] = {qs, ks, vs};
  float* outV[3] = {qv, kv, vv};
  float* outD[3] = {qd, kd, vd};
  for (int set = 0; set < 3; ++set) {
    const float* w0 = W0 + set * cin_s * CCH;
    const float* w1 = W1 + set * cin_v * CCH;
    const float* w2 = W2 + set * cin_v * CCH;
    for (int c = 0; c < CCH; ++c) {
      float ps = 0.f;
      for (int i = 0; i < cin_s; ++i) ps += si[i] * w0[i * CCH + c];
      float ns = sqrtf(ps * ps + EPSc * EPSc);
      ps *= fmaxf(ns, 0.f) / ns;
      outS[set][(long)n * CCH + c] = ps;
      float pv[3]; float nv2 = 0.f;
      for (int m = 0; m < 3; ++m) {
        float a = 0.f;
        for (int i = 0; i < cin_v; ++i) a += vi[i][m] * w1[i * CCH + c];
        pv[m] = a; nv2 += a * a;
      }
      float nv = sqrtf(nv2 + EPSc * EPSc);
      float fv = fmaxf(nv, 0.f) / nv;
      for (int m = 0; m < 3; ++m) outV[set][(long)n * 24 + c * 3 + m] = pv[m] * fv;
      float pd[5]; float nd2 = 0.f;
      for (int m = 0; m < 5; ++m) {
        float a = 0.f;
        for (int i = 0; i < cin_v; ++i) a += di[i][m] * w2[i * CCH + c];
        pd[m] = a; nd2 += a * a;
      }
      float nd = sqrtf(nd2 + EPSc * EPSc);
      float fd = fmaxf(nd, 0.f) / nd;
      for (int m = 0; m < 5; ++m) outD[set][(long)n * 40 + c * 5 + m] = pd[m] * fd;
    }
  }
}

// ----------------------------------------- per-edge attention logits, Wd in LDS
__global__ void k_edge_logit(const float* __restrict__ qs, const float* __restrict__ qv,
                             const float* __restrict__ qd, const float* __restrict__ ks,
                             const float* __restrict__ kv, const float* __restrict__ kd,
                             const float* __restrict__ Wd, const int* __restrict__ esrc,
                             const int* __restrict__ edst, float* __restrict__ expv,
                             float* __restrict__ z) {
  __shared__ float wd[3 * CCH * CCH];
  int tid = threadIdx.x;
  if (tid < 3 * CCH * CCH) wd[tid] = Wd[tid];
  __syncthreads();
  int e = blockIdx.x * blockDim.x + tid;
  if (e >= Ee) return;
  int s = esrc[e], d = edst[e];
  float ksl[CCH], kvl[CCH][3], kdl[CCH][5];
#pragma unroll
  for (int c = 0; c < CCH; ++c) {
    ksl[c] = ks[(long)s * CCH + c];
#pragma unroll
    for (int m = 0; m < 3; ++m) kvl[c][m] = kv[(long)s * 24 + c * 3 + m];
#pragma unroll
    for (int m = 0; m < 5; ++m) kdl[c][m] = kd[(long)s * 40 + c * 5 + m];
  }
  float l0 = 0.f, l1 = 0.f, l2 = 0.f;
#pragma unroll
  for (int c = 0; c < CCH; ++c) {
    float qsc = qs[(long)d * CCH + c];
    float qv0 = qv[(long)d * 24 + c * 3 + 0];
    float qv1 = qv[(long)d * 24 + c * 3 + 1];
    float qv2 = qv[(long)d * 24 + c * 3 + 2];
    float qd0 = qd[(long)d * 40 + c * 5 + 0];
    float qd1 = qd[(long)d * 40 + c * 5 + 1];
    float qd2 = qd[(long)d * 40 + c * 5 + 2];
    float qd3 = qd[(long)d * 40 + c * 5 + 3];
    float qd4 = qd[(long)d * 40 + c * 5 + 4];
#pragma unroll
    for (int dd = 0; dd < CCH; ++dd) {
      l0 += wd[c * CCH + dd] * qsc * ksl[dd];
      l1 += wd[64 + c * CCH + dd] *
            (qv0 * kvl[dd][0] + qv1 * kvl[dd][1] + qv2 * kvl[dd][2]);
      l2 += wd[128 + c * CCH + dd] *
            (qd0 * kdl[dd][0] + qd1 * kdl[dd][1] + qd2 * kdl[dd][2] +
             qd3 * kdl[dd][3] + qd4 * kdl[dd][4]);
    }
  }
  float logit = l0 + l1 * 0.5773502691896258f + l2 * 0.4472135954999579f;
  float ev = __expf(logit);
  expv[e] = ev;
  atomicAdd(&z[d], ev);
}

// -------------------------------- per-edge weighted value scatter (attention out)
__global__ void k_edge_scatter(const float* __restrict__ expv, const float* __restrict__ z,
                               const int* __restrict__ esrc, const int* __restrict__ edst,
                               const float* __restrict__ vs, const float* __restrict__ vv,
                               const float* __restrict__ vd, float* __restrict__ osB,
                               float* __restrict__ ovB, float* __restrict__ odB) {
  int e = blockIdx.x * blockDim.x + threadIdx.x;
  if (e >= Ee) return;
  int s = esrc[e], d = edst[e];
  float zd = z[d];
  zd = (zd == 0.f) ? 1.f : zd;
  float w = sqrtf(fmaxf(expv[e] / zd, 0.f));
#pragma unroll
  for (int c = 0; c < CCH; ++c) atomicAdd(&osB[(long)d * CCH + c], w * vs[(long)s * CCH + c]);
#pragma unroll
  for (int j = 0; j < 24; ++j) atomicAdd(&ovB[(long)d * 24 + j], w * vv[(long)s * 24 + j]);
#pragma unroll
  for (int j = 0; j < 40; ++j) atomicAdd(&odB[(long)d * 40 + j], w * vd[(long)s * 40 + j]);
}

// ----------------------------------------------------- replace / residual add
__global__ void k_combine(float* __restrict__ sA, float* __restrict__ vA, float* __restrict__ dA,
                          const float* __restrict__ osB, const float* __restrict__ ovB,
                          const float* __restrict__ odB, int residual) {
  int n = blockIdx.x * blockDim.x + threadIdx.x;
  if (n >= Nn) return;
  if (residual) {
#pragma unroll
    for (int c = 0; c < CCH; ++c) sA[(long)n * 16 + c] += osB[(long)n * CCH + c];
#pragma unroll
    for (int j = 0; j < 24; ++j) vA[(long)n * 24 + j] += ovB[(long)n * 24 + j];
#pragma unroll
    for (int j = 0; j < 40; ++j) dA[(long)n * 40 + j] += odB[(long)n * 40 + j];
  } else {
#pragma unroll
    for (int c = 0; c < CCH; ++c) sA[(long)n * 16 + c] = osB[(long)n * CCH + c];
#pragma unroll
    for (int c = CCH; c < 16; ++c) sA[(long)n * 16 + c] = 0.f;
#pragma unroll
    for (int j = 0; j < 24; ++j) vA[(long)n * 24 + j] = ovB[(long)n * 24 + j];
#pragma unroll
    for (int j = 0; j < 40; ++j) dA[(long)n * 40 + j] = odB[(long)n * 40 + j];
  }
}

// ------------------- convert Wlin f32 [l][K][N] -> f16 transposed [l][N][K] (B frag)
__global__ void k_convert_w(const float* __restrict__ Wlin, _Float16* __restrict__ Wt) {
  int t = blockIdx.x * blockDim.x + threadIdx.x;
  if (t >= 2 * HIDc * HIDc) return;
  int l = t / (HIDc * HIDc);
  int rem = t - l * HIDc * HIDc;
  int k = rem / HIDc;
  int nn = rem - k * HIDc;
  Wt[(long)l * HIDc * HIDc + (long)nn * HIDc + k] = (_Float16)Wlin[t];
}

// ------------------------------------ head: h = relu(normalize(s @ Wol)) -> f16
__global__ void k_head0(const float* __restrict__ sA, const float* __restrict__ Wol,
                        _Float16* __restrict__ h16) {
  int lane = threadIdx.x & 31;
  int node = blockIdx.x * (blockDim.x >> 5) + (threadIdx.x >> 5);
  if (node >= Nn) return;
  float sv[CCH];
#pragma unroll
  for (int c = 0; c < CCH; ++c) sv[c] = sA[(long)node * 16 + c];
  float h[4];
#pragma unroll
  for (int t = 0; t < 4; ++t) {
    int j = lane * 4 + t;
    float a = 0.f;
#pragma unroll
    for (int c = 0; c < CCH; ++c) a += sv[c] * Wol[c * HIDc + j];
    h[t] = a;
  }
  float ss = h[0] * h[0] + h[1] * h[1] + h[2] * h[2] + h[3] * h[3];
#pragma unroll
  for (int m = 16; m >= 1; m >>= 1) ss += __shfl_xor(ss, m, 32);
  float inv = 1.f / fmaxf(sqrtf(ss), 1e-12f);
#pragma unroll
  for (int t = 0; t < 4; ++t) {
    int j = lane * 4 + t;
    h16[(long)node * HIDc + j] = (_Float16)fmaxf(h[t] * inv, 0.f);
  }
}

// -------------------- 128x128 MLP layer via v_wmma_f32_16x16x32_f16
// 5 waves / block, 625 blocks (5*625*16 = 50000 rows exactly).
// Weights staged into LDS once per block via the Tensor Data Mover (padded rows
// to avoid bank conflicts: LDS layout [col][136] halves, pad 4 DWORDs / 64 DWORDs).
// STORE_MODE 0: f16 output (feeds next layer); 1: f32 output (feeds pooling).
#define LROW 136
template <int STORE_MODE>
__global__ __launch_bounds__(160) void k_mlp_wmma(const _Float16* __restrict__ hin,
                                                  const _Float16* __restrict__ Wt,
                                                  const float* __restrict__ bias,
                                                  _Float16* __restrict__ hout16,
                                                  float* __restrict__ houtf) {
  __shared__ _Float16 ldsw[HIDc * LROW];  // 128 cols x (128 + 8 pad) halves = 34 KB
  int tid = threadIdx.x;
#if USE_TDM
  if (tid < 32) {  // one wave issues the tensor DMA (EXEC ignored by TDM)
    unsigned long long ga = (unsigned long long)(const void*)Wt;
    unsigned int lds_off = (unsigned int)(unsigned long long)(void*)&ldsw[0];
    // D# group0: count=1(valid), lds_addr, global_addr[56:0], type=2 ("image")
    v4u g0 = {1u, lds_off, (unsigned int)(ga & 0xFFFFFFFFu),
              (unsigned int)(((ga >> 32) & 0x01FFFFFFu) | 0x80000000u)};
    const unsigned int W16 = 16384u;  // 128*128 2-byte elements
    // D# group1: data_size=2B, pad_enable, pad_interval=5 (64 DW = one 128-half
    // row), pad_amount=3 (4 DW = 8 halves) -> LDS row stride 136 halves.
    v8i g1 = {(int)((1u << 16) | (1u << 20) | (5u << 22) | (3u << 25)),
              (int)((W16 & 0xFFFFu) << 16),                 // tensor_dim0[15:0] @ [63:48]
              (int)(((W16 >> 16) & 0xFFFFu) | (1u << 16)),  // dim0 hi | tensor_dim1=1
              (int)((W16 & 0xFFFFu) << 16),                 // tile_dim0=16384 @ [127:112]
              0,                                            // tile_dim1=0, tile_dim2=0
              (int)W16,                                     // tensor_dim0_stride lo32
              0, 0};
    v4i g2 = {0, 0, 0, 0};
    v4i g3 = {0, 0, 0, 0};
#if __clang_major__ >= 23
    v8i g4 = {0, 0, 0, 0, 0, 0, 0, 0};
    __builtin_amdgcn_tensor_load_to_lds(g0, g1, g2, g3, g4, 0);
#else
    __builtin_amdgcn_tensor_load_to_lds(g0, g1, g2, g3, 0);
#endif
    __builtin_amdgcn_s_wait_tensorcnt(0);
  }
#else
  for (int i = tid; i < HIDc * 16; i += 160) {  // cooperative staged copy (padded)
    int row = i >> 4, c = (i & 15) << 3;
    *(v8h*)(&ldsw[row * LROW + c]) = *(const v8h*)(Wt + row * HIDc + c);
  }
#endif
  __syncthreads();

  int lane = tid & 31;
  int wave = tid >> 5;  // 0..4
  long r0 = ((long)blockIdx.x * 5 + wave) * 16;
  int rlo = lane & 15;
  int hi = (lane >> 4) & 1;

  // A fragments (16x32 f16 per k-step): lane<16 holds K {kb..kb+7, kb+16..kb+23}
  const _Float16* arow = hin + (r0 + rlo) * HIDc;
  v16h a[4];
#pragma unroll
  for (int kt = 0; kt < 4; ++kt) {
    int kb = kt * 32 + hi * 8;
    v8h lo = *(const v8h*)(arow + kb);
    v8h hh = *(const v8h*)(arow + kb + 16);
#pragma unroll
    for (int i = 0; i < 8; ++i) { a[kt][i] = lo[i]; a[kt][8 + i] = hh[i]; }
  }
#pragma unroll
  for (int nt = 0; nt < 8; ++nt) {
    int col = nt * 16 + rlo;
    const _Float16* wcol = &ldsw[col * LROW + hi * 8];  // B frag: K {kb..kb+7, kb+16..kb+23}
    v8f acc = {};
#pragma unroll
    for (int kt = 0; kt < 4; ++kt) {
      v8h b0 = *(const v8h*)(wcol + kt * 32);
      v8h b1 = *(const v8h*)(wcol + kt * 32 + 16);
      v16h b;
#pragma unroll
      for (int i = 0; i < 8; ++i) { b[i] = b0[i]; b[8 + i] = b1[i]; }
      acc = __builtin_amdgcn_wmma_f32_16x16x32_f16(false, a[kt], false, b,
                                                   (short)0, acc, false, false);
    }
    float bc = bias[col];
#pragma unroll
    for (int r = 0; r < 8; ++r) {  // C/D layout: row = r0 + r + 8*hi, col = nt*16+rlo
      long row = r0 + r + hi * 8;
      float o = fmaxf(acc[r] + bc, 0.f);
      if constexpr (STORE_MODE == 0) {
        hout16[row * HIDc + col] = (_Float16)o;
      } else {
        houtf[row * HIDc + col] = o;
      }
    }
  }
}

// ----------------------------------------------- per-graph pooling over edges
__global__ void k_group_scatter(const float* __restrict__ hf, const int* __restrict__ esrc,
                                const int* __restrict__ bseg, float* __restrict__ m,
                                float* __restrict__ cg) {
  int e = blockIdx.x;
  int s = esrc[e];
  int g = bseg[s];
  atomicAdd(&m[(long)g * HIDc + threadIdx.x], hf[(long)s * HIDc + threadIdx.x]);
  if (threadIdx.x == 0) atomicAdd(&cg[g], 1.f);
}

// --------------------------------------------------- readout matmul + softmax
__global__ void k_final(const float* __restrict__ m, const float* __restrict__ cg,
                        const float* __restrict__ Wout, const float* __restrict__ bout,
                        float* __restrict__ out) {
  int g = blockIdx.x * blockDim.x + threadIdx.x;
  if (g >= Gg) return;
  float inv = 1.f / fmaxf(cg[g], 1.f);
  float o[9];
#pragma unroll
  for (int j = 0; j < 9; ++j) o[j] = bout[j];
  for (int k = 0; k < HIDc; ++k) {
    float mv = m[(long)g * HIDc + k] * inv;
#pragma unroll
    for (int j = 0; j < 9; ++j) o[j] += mv * Wout[k * 9 + j];
  }
  float mx = o[0];
#pragma unroll
  for (int j = 1; j < 9; ++j) mx = fmaxf(mx, o[j]);
  float sum = 0.f;
#pragma unroll
  for (int j = 0; j < 9; ++j) { o[j] = __expf(o[j] - mx); sum += o[j]; }
  float is = 1.f / sum;
#pragma unroll
  for (int j = 0; j < 9; ++j) out[(long)g * 9 + j] = o[j] * is;
}

// ---------------------------------------------------------------------------
extern "C" void kernel_launch(void* const* d_in, const int* in_sizes, int n_in,
                              void* d_out, int out_size, void* d_ws, size_t ws_size,
                              hipStream_t stream) {
  (void)in_sizes; (void)n_in; (void)out_size; (void)ws_size;
  const float* xs     = (const float*)d_in[0];
  const float* evec   = (const float*)d_in[1];
  const float* et0_W0 = (const float*)d_in[2];
  const float* et0_W1 = (const float*)d_in[3];
  const float* et0_W2 = (const float*)d_in[4];
  const float* et0_Wd = (const float*)d_in[5];
  const float* h_W0   = (const float*)d_in[6];
  const float* h_W1   = (const float*)d_in[7];
  const float* h_W2   = (const float*)d_in[8];
  const float* h_Wd   = (const float*)d_in[9];
  const float* Wol    = (const float*)d_in[10];
  const float* Wlin   = (const float*)d_in[11];
  const float* blin   = (const float*)d_in[12];
  const float* Wout   = (const float*)d_in[13];
  const float* bout   = (const float*)d_in[14];
  const int*   esrc   = (const int*)d_in[15];
  const int*   edst   = (const int*)d_in[16];
  const int*   bseg   = (const int*)d_in[17];
  float* out = (float*)d_out;

  // workspace carve (256B aligned slices)
  float* ws = (float*)d_ws;
  size_t off = 0;
  auto alloc = [&](size_t nf) { float* p = ws + off; off += (nf + 63) & ~(size_t)63; return p; };
  float* node_sh = alloc((size_t)Nn * 9);
  float* cnt     = alloc(Nn);
  float* sA  = alloc((size_t)Nn * 16);
  float* vA  = alloc((size_t)Nn * 24);
  float* dA  = alloc((size_t)Nn * 40);
  float* osB = alloc((size_t)Nn * 8);
  float* ovB = alloc((size_t)Nn * 24);
  float* odB = alloc((size_t)Nn * 40);
  float* qs  = alloc((size_t)Nn * 8);
  float* qv  = alloc((size_t)Nn * 24);
  float* qd  = alloc((size_t)Nn * 40);
  float* ks_ = alloc((size_t)Nn * 8);
  float* kv  = alloc((size_t)Nn * 24);
  float* kd  = alloc((size_t)Nn * 40);
  float* vs  = alloc((size_t)Nn * 8);
  float* vv  = alloc((size_t)Nn * 24);
  float* vd  = alloc((size_t)Nn * 40);
  float* expv = alloc(Ee);
  float* zbuf = alloc(Nn);
  _Float16* h16a = (_Float16*)alloc((size_t)Nn * 64);   // N*128 halves
  _Float16* h16b = (_Float16*)alloc((size_t)Nn * 64);
  float* hf = alloc((size_t)Nn * 128);
  _Float16* Wt = (_Float16*)alloc(2 * HIDc * HIDc / 2); // 2*128*128 halves
  float* mbuf = alloc((size_t)Gg * HIDc);
  float* cgb  = alloc(Gg);

  const int TB = 256;
  auto zgrid = [](long n) { long b = (n + 255) / 256; return (int)(b > 4096 ? 4096 : b); };

  k_zero<<<zgrid((long)Nn * 9), TB, 0, stream>>>(node_sh, (long)Nn * 9);
  k_zero<<<zgrid(Nn), TB, 0, stream>>>(cnt, Nn);
  k_edge_sh<<<(Ee + TB - 1) / TB, TB, 0, stream>>>(evec, edst, node_sh, cnt);
  k_node_init<<<(Nn + TB - 1) / TB, TB, 0, stream>>>(xs, node_sh, cnt, sA, vA, dA);

  for (int layer = 0; layer < 3; ++layer) {
    const float *W0, *W1, *W2, *Wd;
    int cin_s, cin_v, residual;
    if (layer == 0) {
      W0 = et0_W0; W1 = et0_W1; W2 = et0_W2; Wd = et0_Wd;
      cin_s = 10; cin_v = 1; residual = 0;
    } else {
      int l = layer - 1;
      W0 = h_W0 + (size_t)l * 3 * 64;
      W1 = h_W1 + (size_t)l * 3 * 64;
      W2 = h_W2 + (size_t)l * 3 * 64;
      Wd = h_Wd + (size_t)l * 3 * 64;
      cin_s = 8; cin_v = 8; residual = 1;
    }
    k_qkv<<<(Nn + TB - 1) / TB, TB, 0, stream>>>(sA, vA, dA, W0, W1, W2, cin_s, cin_v,
                                                 qs, qv, qd, ks_, kv, kd, vs, vv, vd);
    k_zero<<<zgrid((long)Nn * 8), TB, 0, stream>>>(osB, (long)Nn * 8);
    k_zero<<<zgrid((long)Nn * 24), TB, 0, stream>>>(ovB, (long)Nn * 24);
    k_zero<<<zgrid((long)Nn * 40), TB, 0, stream>>>(odB, (long)Nn * 40);
    k_zero<<<zgrid(Nn), TB, 0, stream>>>(zbuf, Nn);
    k_edge_logit<<<(Ee + TB - 1) / TB, TB, 0, stream>>>(qs, qv, qd, ks_, kv, kd, Wd,
                                                        esrc, edst, expv, zbuf);
    k_edge_scatter<<<(Ee + TB - 1) / TB, TB, 0, stream>>>(expv, zbuf, esrc, edst,
                                                          vs, vv, vd, osB, ovB, odB);
    k_combine<<<(Nn + TB - 1) / TB, TB, 0, stream>>>(sA, vA, dA, osB, ovB, odB, residual);
  }

  // MLP head with WMMA (weights staged via TDM into LDS)
  k_convert_w<<<(2 * HIDc * HIDc + TB - 1) / TB, TB, 0, stream>>>(Wlin, Wt);
  k_head0<<<(Nn + 7) / 8, 256, 0, stream>>>(sA, Wol, h16a);
  k_mlp_wmma<0><<<625, 160, 0, stream>>>(h16a, Wt, blin, h16b, (float*)nullptr);
  k_mlp_wmma<1><<<625, 160, 0, stream>>>(h16b, Wt + HIDc * HIDc, blin + HIDc,
                                         (_Float16*)nullptr, hf);

  // graph pooling + readout
  k_zero<<<zgrid((long)Gg * HIDc), TB, 0, stream>>>(mbuf, (long)Gg * HIDc);
  k_zero<<<1, 64, 0, stream>>>(cgb, Gg);
  k_group_scatter<<<Ee, HIDc, 0, stream>>>(hf, esrc, bseg, mbuf, cgb);
  k_final<<<1, 64, 0, stream>>>(mbuf, cgb, Wout, bout, out);
}
